// Bottleneck_SAA_74929999446348
// MI455X (gfx1250) — matrix operand order repaired and checked by reading, not compiled, not run on hardware
//
#include <hip/hip_runtime.h>
#include <hip/hip_bf16.h>
#include <math.h>

typedef __attribute__((ext_vector_type(16))) _Float16 v16h;
typedef __attribute__((ext_vector_type(8)))  _Float16 v8h;
typedef __attribute__((ext_vector_type(8)))  float    v8f;

static constexpr int BB = 4, C2 = 64, HH = 64, WW = 64, NN = HH * WW; // N = 4096
#define BN_EPS 1e-5f

__device__ __forceinline__ float silu_f(float v) {
    return v / (1.f + __expf(-v));
}

// ---------------------------------------------------------------------------
// Conv3x3 (stride 1, pad 1) + BatchNorm(inference) + SiLU, NCHW.
// One thread per output element. FLOP-light vs attention; VALU path is fine.
// ---------------------------------------------------------------------------
template<int IC, int OC>
__global__ void conv_bn_silu(const float* __restrict__ x, const float* __restrict__ w,
                             const float* __restrict__ g, const float* __restrict__ bta,
                             const float* __restrict__ mu, const float* __restrict__ var,
                             float* __restrict__ out) {
    int idx = blockIdx.x * blockDim.x + threadIdx.x;      // ((b*OC + o)*H + h)*W + wq
    if (idx >= BB * OC * NN) return;
    int wq = idx % WW; int t = idx / WW;
    int h  = t % HH;   t /= HH;
    int o  = t % OC;   int b = t / OC;

    const float* xb = x + (size_t)b * IC * NN;
    const float* wo = w + (size_t)o * IC * 9;
    float acc = 0.f;
    for (int i = 0; i < IC; ++i) {
        const float* xi = xb + (size_t)i * NN;
        const float* wi = wo + i * 9;
        #pragma unroll
        for (int kh = 0; kh < 3; ++kh) {
            int hh = h + kh - 1;
            if ((unsigned)hh >= (unsigned)HH) continue;
            const float* xr = xi + hh * WW;
            #pragma unroll
            for (int kw = 0; kw < 3; ++kw) {
                int wc = wq + kw - 1;
                if ((unsigned)wc >= (unsigned)WW) continue;
                acc = fmaf(xr[wc], wi[kh * 3 + kw], acc);
            }
        }
    }
    float sc = g[o] * rsqrtf(var[o] + BN_EPS);
    out[idx] = silu_f(fmaf(acc - mu[o], sc, bta[o]));
}

// ---------------------------------------------------------------------------
// Q/K 1x1 projections: per token, 8-channel outputs, stored f16 [B][N][8]
// (16 bytes per token -> one aligned v8h store, feeds WMMA fragments directly)
// ---------------------------------------------------------------------------
__global__ void qk_proj(const float* __restrict__ y,
                        const float* __restrict__ qw, const float* __restrict__ qb,
                        const float* __restrict__ kw, const float* __restrict__ kb,
                        _Float16* __restrict__ Qh, _Float16* __restrict__ Kh) {
    int idx = blockIdx.x * blockDim.x + threadIdx.x;      // b*N + n
    if (idx >= BB * NN) return;
    int n = idx % NN; int b = idx / NN;
    const float* yb = y + (size_t)b * C2 * NN + n;
    float yv[C2];
    #pragma unroll
    for (int c = 0; c < C2; ++c) yv[c] = yb[(size_t)c * NN];
    v8h qo, ko;
    #pragma unroll
    for (int o = 0; o < 8; ++o) {
        float aq = qb[o], ak = kb[o];
        const float* qr = qw + o * C2;
        const float* kr = kw + o * C2;
        #pragma unroll
        for (int c = 0; c < C2; ++c) {
            aq = fmaf(qr[c], yv[c], aq);
            ak = fmaf(kr[c], yv[c], ak);
        }
        qo[o] = (_Float16)aq; ko[o] = (_Float16)ak;
    }
    *(v8h*)(Qh + (size_t)idx * 8) = qo;
    *(v8h*)(Kh + (size_t)idx * 8) = ko;
}

// ---------------------------------------------------------------------------
// V 1x1 projection: Vh[b][c][n] f16 (channel-major rows load straight into the
// WMMA A-layout of the P·V matmul as contiguous 16B chunks).
// ---------------------------------------------------------------------------
__global__ void v_proj(const float* __restrict__ y, const float* __restrict__ vw,
                       const float* __restrict__ vb, _Float16* __restrict__ Vh) {
    int idx = blockIdx.x * blockDim.x + threadIdx.x;      // ((b*C2 + c)*N + n)
    if (idx >= BB * C2 * NN) return;
    int n = idx % NN; int t = idx / NN;
    int c = t % C2;   int b = t / C2;
    const float* yb = y + (size_t)b * C2 * NN + n;
    const float* wr = vw + c * C2;
    float a = vb[c];
    #pragma unroll 8
    for (int c2 = 0; c2 < C2; ++c2) a = fmaf(wr[c2], yb[(size_t)c2 * NN], a);
    Vh[idx] = (_Float16)a;
}

// ---------------------------------------------------------------------------
// Flash-style PAM attention, one 16-query tile per wave32, 64 keys/iteration.
//   S^T[key][q]  = K_tile(16x8, K-padded to 32) x Q^T     -> 4x v_wmma/iter
//   online softmax over keys (in-register + one xor-16 half-wave shuffle)
//   O^T[c][q]   += V_tile(16x32) x P^T(32x16)             -> 8x v_wmma/iter
// All operand fragments are assembled with branch-free dword selects
// (no EXEC save/restore in the hot loop); the P^T operands need only ONE
// packed xor-16 dword exchange each (4 ds_bpermute per fragment).
// Epilogue fuses out = x + 2*(gamma*O/l + y).
// ---------------------------------------------------------------------------
typedef union { v8h  h; uint4 u; }       H8;
typedef union { v16h h; unsigned u[8]; } H16;

__global__ void __launch_bounds__(128, 1)
pam_attn(const _Float16* __restrict__ Qh, const _Float16* __restrict__ Kh,
         const _Float16* __restrict__ Vh, const float* __restrict__ y,
         const float* __restrict__ x, const float* __restrict__ gamma_p,
         float* __restrict__ out) {
    const int lane = threadIdx.x & 31;
    const int wave = threadIdx.x >> 5;
    const int gw   = blockIdx.x * 4 + wave;   // 0 .. B*(N/16)-1 = 1023
    const int b    = gw >> 8;                 // / (N/16)
    const int qt   = gw & 255;
    const bool lo  = lane < 16;
    const int  l16 = lane & 15;

    const _Float16* Kb = Kh + (size_t)b * NN * 8;
    const _Float16* Vb = Vh + (size_t)b * C2 * NN;

    // Loop-invariant Q^T operand (B-matrix, 32x16): lane = query column,
    // lo lanes hold channels 0..15 (first 8 real), hi lanes channels 16..31 (0).
    // Unconditional load (hi lanes read their partner's address) + dword select.
    H16 qf;
    {
        uint4 qv = *(const uint4*)(Qh + ((size_t)b * NN + qt * 16 + l16) * 8);
        qf.u[0] = lo ? qv.x : 0u;  qf.u[1] = lo ? qv.y : 0u;
        qf.u[2] = lo ? qv.z : 0u;  qf.u[3] = lo ? qv.w : 0u;
        qf.u[4] = 0u; qf.u[5] = 0u; qf.u[6] = 0u; qf.u[7] = 0u;
    }

    float m_run = -1e30f, l_run = 0.f;
    v8f zc = {};
    v8f Ofr[4];
    #pragma unroll
    for (int ct = 0; ct < 4; ++ct) Ofr[ct] = zc;

    for (int kb = 0; kb < NN; kb += 64) {
        // ---- 4 S^T fragments: lane = query, VGPR r = key (lo: r / hi: r+8) ----
        v8f st[4];
        #pragma unroll
        for (int t = 0; t < 4; ++t) {
            uint4 kv = *(const uint4*)(Kb + (size_t)(kb + t * 16 + l16) * 8);
            H16 ka;
            ka.u[0] = lo ? kv.x : 0u;  ka.u[1] = lo ? kv.y : 0u;
            ka.u[2] = lo ? kv.z : 0u;  ka.u[3] = lo ? kv.w : 0u;
            ka.u[4] = 0u; ka.u[5] = 0u; ka.u[6] = 0u; ka.u[7] = 0u;
            st[t] = __builtin_amdgcn_wmma_f32_16x16x32_f16(false, ka.h, false, qf.h,
                                                           (short)0, zc, false, false);
        }

        // ---- online softmax over the 64 keys of this iteration ----
        float tmax = -1e30f;
        #pragma unroll
        for (int t = 0; t < 4; ++t)
            #pragma unroll
            for (int r = 0; r < 8; ++r) tmax = fmaxf(tmax, st[t][r]);
        tmax = fmaxf(tmax, __shfl_xor(tmax, 16, 32));
        float m_new = fmaxf(m_run, tmax);
        float alpha = __expf(m_run - m_new);

        float p[4][8], lsum = 0.f;
        #pragma unroll
        for (int t = 0; t < 4; ++t)
            #pragma unroll
            for (int r = 0; r < 8; ++r) {
                p[t][r] = __expf(st[t][r] - m_new);
                lsum += p[t][r];
            }
        lsum += __shfl_xor(lsum, 16, 32);
        l_run = fmaf(l_run, alpha, lsum);
        m_run = m_new;

        // ---- pack P to f16, then build the two P^T B-operands with ONE
        //      packed xor-16 dword exchange each (4 ds_bpermute / fragment) ----
        H8 pk[4];
        #pragma unroll
        for (int t = 0; t < 4; ++t)
            #pragma unroll
            for (int r = 0; r < 8; ++r) pk[t].h[r] = (_Float16)p[t][r];

        H16 bp[2];
        #pragma unroll
        for (int g2 = 0; g2 < 2; ++g2) {
            const uint4 a = pk[2 * g2].u;      // keys sub-tile, rows 0..7  per lane
            const uint4 c = pk[2 * g2 + 1].u;  // keys sub-tile, rows of 2nd 16-tile
            uint4 send;                        // what the xor-16 partner needs
            send.x = lo ? c.x : a.x;  send.y = lo ? c.y : a.y;
            send.z = lo ? c.z : a.z;  send.w = lo ? c.w : a.w;
            uint4 recv;
            recv.x = (unsigned)__shfl_xor((int)send.x, 16, 32);
            recv.y = (unsigned)__shfl_xor((int)send.y, 16, 32);
            recv.z = (unsigned)__shfl_xor((int)send.z, 16, 32);
            recv.w = (unsigned)__shfl_xor((int)send.w, 16, 32);
            // lo lane: K=0..15  -> {own loTile, partner loTile}
            // hi lane: K=16..31 -> {partner hiTile, own hiTile}
            bp[g2].u[0] = lo ? a.x : recv.x;  bp[g2].u[1] = lo ? a.y : recv.y;
            bp[g2].u[2] = lo ? a.z : recv.z;  bp[g2].u[3] = lo ? a.w : recv.w;
            bp[g2].u[4] = lo ? recv.x : c.x;  bp[g2].u[5] = lo ? recv.y : c.y;
            bp[g2].u[6] = lo ? recv.z : c.z;  bp[g2].u[7] = lo ? recv.w : c.w;
        }

        // ---- O^T += V x P^T : 4 channel tiles x 2 key sub-tiles of 32 ----
        #pragma unroll
        for (int ct = 0; ct < 4; ++ct) {
            const _Float16* vs = Vb + (size_t)(ct * 16 + l16) * NN + kb + (lo ? 0 : 16);
            uint4 v0 = *(const uint4*)(vs);
            uint4 v1 = *(const uint4*)(vs + 8);
            uint4 v2 = *(const uint4*)(vs + 32);
            uint4 v3 = *(const uint4*)(vs + 40);
            H16 va0, va1;
            va0.u[0] = v0.x; va0.u[1] = v0.y; va0.u[2] = v0.z; va0.u[3] = v0.w;
            va0.u[4] = v1.x; va0.u[5] = v1.y; va0.u[6] = v1.z; va0.u[7] = v1.w;
            va1.u[0] = v2.x; va1.u[1] = v2.y; va1.u[2] = v2.z; va1.u[3] = v2.w;
            va1.u[4] = v3.x; va1.u[5] = v3.y; va1.u[6] = v3.z; va1.u[7] = v3.w;
            v8f cacc = Ofr[ct];
            #pragma unroll
            for (int r = 0; r < 8; ++r) cacc[r] *= alpha;   // flash rescale
            v8f mid = __builtin_amdgcn_wmma_f32_16x16x32_f16(false, va0.h, false, bp[0].h,
                                                             (short)0, cacc, false, false);
            Ofr[ct] = __builtin_amdgcn_wmma_f32_16x16x32_f16(false, va1.h, false, bp[1].h,
                                                             (short)0, mid, false, false);
        }
        if (kb + 64 < NN)
            __builtin_prefetch(Kb + (size_t)(kb + 64 + l16) * 8, 0, 0);
    }

    // ---- epilogue: out = x + 2*(gamma*o + y), o = O / l ----
    float inv   = 1.f / l_run;
    float gamma = gamma_p[0];
    const float* xb = x + (size_t)b * C2 * NN;
    const float* yb = y + (size_t)b * C2 * NN;
    float*       ob = out + (size_t)b * C2 * NN;
    int nq = qt * 16 + l16;
    #pragma unroll
    for (int ct = 0; ct < 4; ++ct) {
        #pragma unroll
        for (int r = 0; r < 8; ++r) {
            int c = ct * 16 + r + (lo ? 0 : 8);
            size_t ix = (size_t)c * NN + nq;
            float o = Ofr[ct][r] * inv;
            ob[ix] = xb[ix] + 2.f * fmaf(gamma, o, yb[ix]);
        }
    }
}

// ---------------------------------------------------------------------------
extern "C" void kernel_launch(void* const* d_in, const int* in_sizes, int n_in,
                              void* d_out, int out_size, void* d_ws, size_t ws_size,
                              hipStream_t stream) {
    (void)in_sizes; (void)n_in; (void)out_size; (void)ws_size;

    const float* x     = (const float*)d_in[0];
    const float* cv1_w = (const float*)d_in[1];
    const float* bn1_g = (const float*)d_in[2];
    const float* bn1_b = (const float*)d_in[3];
    const float* bn1_m = (const float*)d_in[4];
    const float* bn1_v = (const float*)d_in[5];
    const float* cv2_w = (const float*)d_in[6];
    const float* bn2_g = (const float*)d_in[7];
    const float* bn2_b = (const float*)d_in[8];
    const float* bn2_m = (const float*)d_in[9];
    const float* bn2_v = (const float*)d_in[10];
    const float* q_w   = (const float*)d_in[11];
    const float* q_b   = (const float*)d_in[12];
    const float* k_w   = (const float*)d_in[13];
    const float* k_b   = (const float*)d_in[14];
    const float* v_w   = (const float*)d_in[15];
    const float* v_b   = (const float*)d_in[16];
    const float* gamma = (const float*)d_in[17];
    float* out = (float*)d_out;

    // workspace layout (bytes): y1 [0,2MB) | y [2MB,6MB) | Qh | Kh | Vh  (~8.5MB)
    char* ws = (char*)d_ws;
    float*    y1 = (float*)(ws);
    float*    y  = (float*)(ws + (size_t)(2u << 20));
    _Float16* Qh = (_Float16*)(ws + (size_t)(6u << 20));
    _Float16* Kh = (_Float16*)(ws + (size_t)(6u << 20) + 262144);
    _Float16* Vh = (_Float16*)(ws + (size_t)(6u << 20) + 524288);

    conv_bn_silu<64, 32><<<(BB * 32 * NN) / 256, 256, 0, stream>>>(
        x, cv1_w, bn1_g, bn1_b, bn1_m, bn1_v, y1);
    conv_bn_silu<32, 64><<<(BB * 64 * NN) / 256, 256, 0, stream>>>(
        y1, cv2_w, bn2_g, bn2_b, bn2_m, bn2_v, y);
    qk_proj<<<(BB * NN) / 256, 256, 0, stream>>>(y, q_w, q_b, k_w, k_b, Qh, Kh);
    v_proj<<<(BB * C2 * NN) / 256, 256, 0, stream>>>(y, v_w, v_b, Vh);
    // B*(N/16) = 1024 waves, 4 waves (128 threads) per block
    pam_attn<<<256, 128, 0, stream>>>(Qh, Kh, Vh, y, x, gamma, out);
}
